// Model_52888227283552
// MI455X (gfx1250) — matrix operand order repaired and checked by reading, not compiled, not run on hardware
//
#include <hip/hip_runtime.h>

// ---------------- problem constants ----------------
constexpr int kH  = 50;     // hidden
constexpr int kHP = 64;     // padded hidden (per gate, N direction)
constexpr int kI  = 4;      // input size
constexpr int kT  = 512;    // timesteps
constexpr int kB  = 4096;   // batch
constexpr int kO  = 121;    // output size
constexpr int kNB = 256;    // gate columns = 4 gates * kHP
constexpr int kKB = 96;     // augmented K: 64 (h,padded) + 4 (x) + 1 (bias) + 27 zero
constexpr int kTB = 16;     // batch rows per workgroup (WMMA M)
constexpr int kAS = 88;     // h-buffer row stride in f16 (16B-aligned, conflict-light)

typedef __attribute__((ext_vector_type(16))) _Float16 v16h;
typedef __attribute__((ext_vector_type(8)))  _Float16 v8h;
typedef __attribute__((ext_vector_type(4)))  _Float16 v4h;
typedef __attribute__((ext_vector_type(8)))  float    v8f;

// ---- fast transcendentals: gfx1250 native v_tanh_f32 everywhere ----
__device__ __forceinline__ float tanhfast(float x) {
#if __has_builtin(__builtin_amdgcn_tanhf)
    return __builtin_amdgcn_tanhf(x);       // single v_tanh_f32
#else
    float e = __expf(2.0f * x);
#if __has_builtin(__builtin_amdgcn_rcpf)
    return 1.0f - 2.0f * __builtin_amdgcn_rcpf(e + 1.0f);
#else
    return 1.0f - 2.0f / (e + 1.0f);
#endif
#endif
}
// sigmoid(x) = 0.5*tanh(x/2) + 0.5 : mul -> v_tanh -> fma (one transcendental)
__device__ __forceinline__ float sigf(float x) {
    return __builtin_fmaf(0.5f, tanhfast(0.5f * x), 0.5f);
}

// A-fragment (16-bit A 16x32 layout): per lane, elements 0-7 at p, 8-15 at p+16
__device__ __forceinline__ v16h ldA(const _Float16* p) {
    v8h lo = *(const v8h*)p;
    v8h hi = *(const v8h*)(p + 16);
    return __builtin_shufflevector(lo, hi, 0,1,2,3,4,5,6,7,8,9,10,11,12,13,14,15);
}
// B-fragment: per lane, elements 0-7 at p, 8-15 at p+16 (B stored [N][K] row-major)
__device__ __forceinline__ v16h ldB(const _Float16* p) {
    v8h lo = *(const v8h*)p;
    v8h hi = *(const v8h*)(p + 16);
    return __builtin_shufflevector(lo, hi, 0,1,2,3,4,5,6,7,8,9,10,11,12,13,14,15);
}

__global__ __launch_bounds__(128) void lstm_wmma_kernel(
    const float* __restrict__ x,      // [B, T, I]
    const float* __restrict__ w_ih,   // [4H, I]
    const float* __restrict__ w_hh,   // [4H, H]
    const float* __restrict__ b_ih,   // [4H]
    const float* __restrict__ b_hh,   // [4H]
    const float* __restrict__ w_out,  // [O, H]
    const float* __restrict__ b_out,  // [O]
    float* __restrict__ out)          // [B, O]
{
    extern __shared__ _Float16 smem[];
    _Float16* sW = smem;                       // [kNB][kKB]  augmented weights (f16)
    _Float16* sX = sW + kNB * kKB;             // [kT][kTB][kI] staged inputs (f16)
    _Float16* sA = sX + kT * kTB * kI;         // [2][kTB][kAS] ping-pong h buffers

    const int tid  = threadIdx.x;
    const int lane = tid & 31;
    const int wv   = tid >> 5;      // wave 0..3
    const int lhi  = lane >> 4;     // 0/1 (half-wave)
    const int llo  = lane & 15;     // 0..15
    const int base = blockIdx.x * kTB;

    // ---- build augmented weight matrix in LDS: row n = gate unit, K contiguous ----
    for (int n = tid; n < kNB; n += 128) {
        int g = n >> 6;             // gate (i,f,g,o)
        int u = n & 63;             // hidden unit within gate (padded to 64)
        _Float16* row = sW + n * kKB;
        if (u < kH) {
            int wr = g * kH + u;    // row in the [4H, *] weight matrices
            for (int k = 0; k < kH; ++k)        row[k] = (_Float16)w_hh[wr * kH + k];
            for (int k = kH; k < kHP; ++k)      row[k] = (_Float16)0.0f;
            for (int k = 0; k < kI; ++k)        row[kHP + k] = (_Float16)w_ih[wr * kI + k];
            row[kHP + kI] = (_Float16)(b_ih[wr] + b_hh[wr]);
            for (int k = kHP + kI + 1; k < kKB; ++k) row[k] = (_Float16)0.0f;
        } else {
            for (int k = 0; k < kKB; ++k)       row[k] = (_Float16)0.0f;
        }
    }

    // ---- stage this WG's x rows for ALL timesteps into LDS (coalesced float4 reads) ----
    for (int idx = tid; idx < kTB * kT; idx += 128) {
        int m = idx / kT;
        int t = idx % kT;
        float4 v = *(const float4*)(x + ((size_t)(base + m) * kT + t) * kI);
        _Float16* d = sX + (t * kTB + m) * kI;
        d[0] = (_Float16)v.x; d[1] = (_Float16)v.y;
        d[2] = (_Float16)v.z; d[3] = (_Float16)v.w;
    }

    // ---- zero both h ping-pong buffers (h0 = 0; padding columns stay 0 forever) ----
    for (int i = tid; i < 2 * kTB * kAS; i += 128) sA[i] = (_Float16)0.0f;
    __syncthreads();

    // ---- persistent B fragments in VGPRs: 4 tiles (one per gate) x 3 K-chunks ----
    // wave w owns N-tiles {w, w+4, w+8, w+12} -> gate j at hidden unit u = 16*w + llo
    v16h bf[4][3];
    #pragma unroll
    for (int j = 0; j < 4; ++j) {
        int n = (wv + 4 * j) * 16 + llo;
        #pragma unroll
        for (int c = 0; c < 3; ++c)
            bf[j][c] = ldB(sW + n * kKB + c * 32 + lhi * 8);
    }

    const int u = wv * 16 + llo;    // hidden unit this lane owns
    float cst[8];
    #pragma unroll
    for (int r = 0; r < 8; ++r) cst[r] = 0.0f;

    // constant part of the x/bias A-fragment: zeros + bias-one lane (K=68)
    v16h a2;
    #pragma unroll
    for (int i = 0; i < 16; ++i) a2[i] = (_Float16)0.0f;
    if (lhi == 0) a2[4] = (_Float16)1.0f;

    for (int t = 0; t < kT; ++t) {
        const _Float16* cur = sA + (t & 1) * (kTB * kAS);        // read h_t
        _Float16*       nxt = sA + ((t + 1) & 1) * (kTB * kAS);  // write h_{t+1}

        // A fragments: chunks 0,1 from h buffer; chunk 2 (x | 1 | 0) in registers
        const _Float16* arow = cur + llo * kAS;                  // M = llo
        v16h a0 = ldA(arow + 0  + lhi * 8);
        v16h a1 = ldA(arow + 32 + lhi * 8);
        if (lhi == 0) {                                // K = 64..71 live only in low half
            v4h xv = *(const v4h*)(sX + (t * kTB + llo) * kI);
            a2[0] = xv[0]; a2[1] = xv[1]; a2[2] = xv[2]; a2[3] = xv[3];
        }

        v8f acc[4];
        #pragma unroll
        for (int j = 0; j < 4; ++j) {
            v8f z = {};
            z = __builtin_amdgcn_wmma_f32_16x16x32_f16(false, a0, false, bf[j][0],
                                                       (short)0, z, false, false);
            z = __builtin_amdgcn_wmma_f32_16x16x32_f16(false, a1, false, bf[j][1],
                                                       (short)0, z, false, false);
            z = __builtin_amdgcn_wmma_f32_16x16x32_f16(false, a2, false, bf[j][2],
                                                       (short)0, z, false, false);
            acc[j] = z;
        }

        // gate nonlinearities + state update; lane holds (M = r + 8*lhi, u) for r=0..7
        #pragma unroll
        for (int r = 0; r < 8; ++r) {
            float ig = sigf(acc[0][r]);
            float fg = sigf(acc[1][r]);
            float gg = tanhfast(acc[2][r]);
            float og = sigf(acc[3][r]);
            float c2 = fg * cst[r] + ig * gg;
            cst[r] = c2;
            float hh = og * tanhfast(c2);
            if (u < kH) {
                int M = r + 8 * lhi;
                nxt[M * kAS + u] = (_Float16)hh;
            }
        }
        __syncthreads();   // h_{t+1} fully written before next step reads it
    }

    // ---- final projection: out = h_last @ W_out^T + b_out ----
    // after t = kT-1, the last write went to buffer (kT & 1) == 0
    const _Float16* hfin = sA;
    for (int idx = tid; idx < kTB * kO; idx += 128) {
        int m = idx / kO;
        int o = idx % kO;
        float s = b_out[o];
        for (int k = 0; k < kH; ++k)
            s += (float)hfin[m * kAS + k] * w_out[o * kH + k];
        out[(size_t)(base + m) * kO + o] = s;
    }
}

extern "C" void kernel_launch(void* const* d_in, const int* in_sizes, int n_in,
                              void* d_out, int out_size, void* d_ws, size_t ws_size,
                              hipStream_t stream) {
    (void)in_sizes; (void)n_in; (void)out_size; (void)d_ws; (void)ws_size;
    const float* x    = (const float*)d_in[0];
    const float* wih  = (const float*)d_in[1];
    const float* whh  = (const float*)d_in[2];
    const float* bih  = (const float*)d_in[3];
    const float* bhh  = (const float*)d_in[4];
    const float* wout = (const float*)d_in[5];
    const float* bout = (const float*)d_in[6];
    float* out = (float*)d_out;

    constexpr size_t smem_bytes =
        (size_t)(kNB * kKB + kT * kTB * kI + 2 * kTB * kAS) * 2;  // f16 elements
    static_assert(smem_bytes < 320u * 1024u, "LDS budget");

    // gfx1250 WGP has 320KB LDS; opt in beyond the 64KB default.
    (void)hipFuncSetAttribute((const void*)lstm_wmma_kernel,
                              hipFuncAttributeMaxDynamicSharedMemorySize,
                              (int)smem_bytes);

    lstm_wmma_kernel<<<kB / kTB, 128, smem_bytes, stream>>>(
        x, wih, whh, bih, bhh, wout, bout, out);
}
